// Attention_70308614636109
// MI455X (gfx1250) — matrix operand order repaired and checked by reading, not compiled, not run on hardware
//
#include <hip/hip_runtime.h>
#include <hip/hip_bf16.h>

// ---------------------------------------------------------------------------
// MI455X (gfx1250) attention block:
//   LN -> QKV GEMM -> flash attention (16 heads, d=64) -> out GEMM
// All matmuls on v_wmma_f32_16x16x32_bf16 (wave32), f32 accumulate.
// Double-buffered LDS staging, global-load / lds-store split so WMMA hides
// global latency; transpose packing done in registers with b32 LDS traffic.
// ---------------------------------------------------------------------------

typedef __attribute__((ext_vector_type(16))) __bf16 v16bf;
typedef __attribute__((ext_vector_type(8)))  float  v8f;

#define DIMX   1024
#define NSEQ   2048
#define BATCH  4
#define HEADS  16
#define HDIM   64
#define ROWS   (BATCH * NSEQ)     // 8192
#define NQKV   (3 * DIMX)         // 3072

__device__ __forceinline__ unsigned short f2bf(float f) {
    unsigned int u = __float_as_uint(f);
    unsigned int r = (u + 0x7FFFu + ((u >> 16) & 1u)) >> 16;   // RNE
    return (unsigned short)r;
}

__device__ __forceinline__ v8f wmma_bf16(v16bf a, v16bf b, v8f c) {
    return __builtin_amdgcn_wmma_f32_16x16x32_bf16(
        false, a, false, b, (short)0, c, false, false);
}

// A-matrix (16x32 bf16) per-lane K map (ISA 7.12.2 table).
__device__ __forceinline__ int kpairA(int v, int hi) {
    return ((v & 4) << 2) + (hi << 3) + ((v & 3) << 1);
}
// B-matrix (32x16 bf16): same per-lane K map as A (ISA 7.12.2 table).
__device__ __forceinline__ int kpairB(int v, int hi) {
    return ((v & 4) << 2) + (hi << 3) + ((v & 3) << 1);
}

__device__ __forceinline__ v16bf load_afrag(const unsigned short* base, int stride,
                                            int row, int kbase, int hi) {
    union { v16bf v; unsigned int u[8]; } f;
#pragma unroll
    for (int i = 0; i < 8; ++i)
        f.u[i] = *(const unsigned int*)(base + (size_t)row * stride + kbase + kpairA(i, hi));
    return f.v;
}

__device__ __forceinline__ v16bf load_bfrag(const unsigned short* base, int stride,
                                            int n, int kbase, int hi) {
    union { v16bf v; unsigned int u[8]; } f;
#pragma unroll
    for (int i = 0; i < 8; ++i)
        f.u[i] = *(const unsigned int*)(base + (size_t)n * stride + kbase + kpairB(i, hi));
    return f.v;
}

__device__ __forceinline__ v8f zero8() {
    v8f z;
#pragma unroll
    for (int i = 0; i < 8; ++i) z[i] = 0.0f;
    return z;
}

// ---------------------------------------------------------------------------
// Kernel 1: LayerNorm (f32 in) -> bf16 out. One block per row of 1024.
// ---------------------------------------------------------------------------
__global__ __launch_bounds__(256) void ln_bf16_kernel(
    const float* __restrict__ x, const float* __restrict__ gamma,
    const float* __restrict__ beta, unsigned short* __restrict__ xn)
{
    int row = blockIdx.x;
    int tid = threadIdx.x;
    int lane = tid & 31, wave = tid >> 5;
    const float* xr = x + (size_t)row * DIMX;

    float v[4], s = 0.0f, s2 = 0.0f;
#pragma unroll
    for (int i = 0; i < 4; ++i) {
        v[i] = xr[tid + i * 256];
        s += v[i];
        s2 += v[i] * v[i];
    }
#pragma unroll
    for (int m = 1; m < 32; m <<= 1) {
        s  += __shfl_xor(s,  m, 32);
        s2 += __shfl_xor(s2, m, 32);
    }
    __shared__ float red[16];
    if (lane == 0) { red[wave] = s; red[8 + wave] = s2; }
    __syncthreads();
    float ts = 0.0f, ts2 = 0.0f;
#pragma unroll
    for (int w = 0; w < 8; ++w) { ts += red[w]; ts2 += red[8 + w]; }
    float mu  = ts * (1.0f / DIMX);
    float var = ts2 * (1.0f / DIMX) - mu * mu;
    float rstd = rsqrtf(var + 1e-5f);
#pragma unroll
    for (int i = 0; i < 4; ++i) {
        int c = tid + i * 256;
        float y = (v[i] - mu) * rstd * gamma[c] + beta[c];
        xn[(size_t)row * DIMX + c] = f2bf(y);
    }
}

// ---------------------------------------------------------------------------
// Kernel 2: f32 -> bf16 elementwise (weights)
// ---------------------------------------------------------------------------
__global__ __launch_bounds__(256) void cvt_bf16_kernel(
    const float* __restrict__ src, unsigned short* __restrict__ dst, int n)
{
    int i = blockIdx.x * 256 + threadIdx.x;
    if (i < n) dst[i] = f2bf(src[i]);
}

// ---------------------------------------------------------------------------
// Kernel 3: bf16 GEMM  C[M,N] = A[M,K] * B[K,N]
//   block tile 128x128, K-step 32, double-buffered LDS, 8 waves,
//   each wave 32x64 (2 A-frags x 4 B-frags = 8 WMMA / step).
//   mode 0: scatter into Q(x0.125)/K/V [bh, n, d] bf16
//   mode 1: write f32 to outF [M,N]
// ---------------------------------------------------------------------------
__global__ __launch_bounds__(256) void gemm_bf16_kernel(
    const unsigned short* __restrict__ A, const unsigned short* __restrict__ B,
    int M, int N, int K, int mode,
    float* __restrict__ outF,
    unsigned short* __restrict__ Qg, unsigned short* __restrict__ Kg,
    unsigned short* __restrict__ Vg)
{
    __shared__ unsigned short ldsA[2][128][32];    // [m][k]
    __shared__ unsigned short ldsBT[2][128][32];   // [n][k]  (transposed)

    int tid  = threadIdx.x;
    int lane = tid & 31, wave = tid >> 5;
    int hi = (lane >> 4) & 1, ln = lane & 15;
    int wm = wave & 3, wn = wave >> 2;            // 4 x 2 wave grid
    int m0 = blockIdx.y * 128;
    int n0 = blockIdx.x * 128;

    v8f acc[2][4];
#pragma unroll
    for (int i = 0; i < 2; ++i)
#pragma unroll
        for (int j = 0; j < 4; ++j) acc[i][j] = zero8();

    // staging registers: A 4 x b32, B 4 x (2 rows) x b32
    unsigned int ra[4], rb0[4], rb1[4];

    auto stage_load = [&](int kk) {
#pragma unroll
        for (int j = 0; j < 4; ++j) {              // A tile: 1024 b32
            int idx = tid + j * 256;
            int row = idx >> 4, c2 = idx & 15;
            ra[j] = *(const unsigned int*)(A + (size_t)(m0 + row) * K + kk + c2 * 2);
        }
#pragma unroll
        for (int j = 0; j < 4; ++j) {              // B tile: 1024 pair-reads
            int idx = tid + j * 256;
            int kp = idx >> 6;                     // 0..15 -> k = 2*kp
            int n2 = idx & 63;                     // 0..63 -> n = 2*n2
            const unsigned short* bp = B + (size_t)(kk + 2 * kp) * N + n0 + 2 * n2;
            rb0[j] = *(const unsigned int*)(bp);
            rb1[j] = *(const unsigned int*)(bp + N);
        }
        if (kk + 64 < K) {                         // pull K+64 tile toward L2
            int row = tid >> 1;
            __builtin_prefetch(A + (size_t)(m0 + row) * K + kk + 64, 0, 1);
            __builtin_prefetch(B + (size_t)(kk + 64 + (tid & 31)) * N + n0 + ((tid >> 5) << 4), 0, 1);
        }
    };
    auto stage_store = [&](int buf) {
        unsigned int* ldsAu = (unsigned int*)&ldsA[buf][0][0];
#pragma unroll
        for (int j = 0; j < 4; ++j) ldsAu[tid + j * 256] = ra[j];
#pragma unroll
        for (int j = 0; j < 4; ++j) {
            int idx = tid + j * 256;
            int kp = idx >> 6, n2 = idx & 63;
            unsigned int lo = (rb0[j] & 0xffffu) | (rb1[j] << 16);
            unsigned int hh = (rb0[j] >> 16) | (rb1[j] & 0xffff0000u);
            *(unsigned int*)&ldsBT[buf][2 * n2][2 * kp]     = lo;
            *(unsigned int*)&ldsBT[buf][2 * n2 + 1][2 * kp] = hh;
        }
    };

    stage_load(0);
    stage_store(0);
    __syncthreads();

    int cur = 0;
    for (int kk = 0; kk < K; kk += 32) {
        bool more = (kk + 32) < K;
        if (more) stage_load(kk + 32);             // global loads in flight

        const unsigned short* As = &ldsA[cur][0][0];
        const unsigned short* Bs = &ldsBT[cur][0][0];
        v16bf a0 = load_afrag(As, 32, wm * 32 + ln,      0, hi);
        v16bf a1 = load_afrag(As, 32, wm * 32 + 16 + ln, 0, hi);
#pragma unroll
        for (int sn = 0; sn < 4; ++sn) {
            v16bf b = load_bfrag(Bs, 32, wn * 64 + sn * 16 + ln, 0, hi);
            acc[0][sn] = wmma_bf16(a0, b, acc[0][sn]);
            acc[1][sn] = wmma_bf16(a1, b, acc[1][sn]);
        }

        if (more) stage_store(cur ^ 1);            // waits on loads, fills other buf
        cur ^= 1;
        __syncthreads();
    }

#pragma unroll
    for (int sm = 0; sm < 2; ++sm) {
#pragma unroll
        for (int sn = 0; sn < 4; ++sn) {
#pragma unroll
            for (int r = 0; r < 8; ++r) {
                int row = m0 + wm * 32 + sm * 16 + r + hi * 8;
                int col = n0 + wn * 64 + sn * 16 + ln;
                float val = acc[sm][sn][r];
                if (mode == 1) {
                    outF[(size_t)row * N + col] = val;
                } else {
                    int which = col >> 10, inner = col & 1023;
                    int head = inner >> 6, dd = inner & 63;
                    int b = row >> 11, i = row & 2047;
                    size_t off = (((size_t)(b * HEADS + head)) * NSEQ + i) * HDIM + dd;
                    if (which == 0)      Qg[off] = f2bf(val * 0.125f);   // SCALE
                    else if (which == 1) Kg[off] = f2bf(val);
                    else                 Vg[off] = f2bf(val);
                }
            }
        }
    }
}

// ---------------------------------------------------------------------------
// Kernel 4: flash attention. grid = (N/128, B*H). Each wave owns 16 q-rows.
//   S = Q*K^T (Q pre-scaled), online softmax, O += P*V. Double-buffered K/V.
// ---------------------------------------------------------------------------
__global__ __launch_bounds__(256) void attn_kernel(
    const unsigned short* __restrict__ Qg, const unsigned short* __restrict__ Kg,
    const unsigned short* __restrict__ Vg, unsigned short* __restrict__ Og)
{
    __shared__ unsigned short ldsK[2][32][64];    // [key][d]  (k-contig for S B-frags)
    __shared__ unsigned short ldsVt[2][64][32];   // [d][key]  (key-contig for PV B-frags)
    __shared__ unsigned short ldsP[128][32];      // per-wave 16x32 P strip

    int tid  = threadIdx.x;
    int lane = tid & 31, wave = tid >> 5;
    int hi = (lane >> 4) & 1, ln = lane & 15;
    int bh = blockIdx.y;                 // b*16 + h
    int h = bh & (HEADS - 1), b = bh >> 4;
    int q0 = blockIdx.x * 128;

    const unsigned short* Qbh = Qg + (size_t)bh * NSEQ * HDIM;
    const unsigned short* Kbh = Kg + (size_t)bh * NSEQ * HDIM;
    const unsigned short* Vbh = Vg + (size_t)bh * NSEQ * HDIM;

    int qrow = q0 + wave * 16 + ln;
    v16bf qa0 = load_afrag(Qbh, HDIM, qrow,  0, hi);
    v16bf qa1 = load_afrag(Qbh, HDIM, qrow, 32, hi);

    v8f o[4];
#pragma unroll
    for (int dt = 0; dt < 4; ++dt) o[dt] = zero8();
    float mrow[8], lrow[8];
#pragma unroll
    for (int r = 0; r < 8; ++r) { mrow[r] = -1e30f; lrow[r] = 0.0f; }

    unsigned int rk[4], rv0[2], rv1[2];
    auto stage_load = [&](int j) {
#pragma unroll
        for (int t = 0; t < 4; ++t) {              // K tile: 1024 b32
            int idx = tid + t * 256;
            int key = idx >> 5, d2 = idx & 31;
            rk[t] = *(const unsigned int*)(Kbh + (size_t)(j + key) * HDIM + d2 * 2);
        }
#pragma unroll
        for (int t = 0; t < 2; ++t) {              // V tile: 512 pair-reads
            int idx = tid + t * 256;
            int kp = idx >> 5;                     // 0..15 -> key = 2*kp
            int d2 = idx & 31;                     // 0..31 -> d = 2*d2
            const unsigned short* vp = Vbh + (size_t)(j + 2 * kp) * HDIM + 2 * d2;
            rv0[t] = *(const unsigned int*)(vp);
            rv1[t] = *(const unsigned int*)(vp + HDIM);
        }
    };
    auto stage_store = [&](int buf) {
        unsigned int* ldsKu = (unsigned int*)&ldsK[buf][0][0];
#pragma unroll
        for (int t = 0; t < 4; ++t) ldsKu[tid + t * 256] = rk[t];
#pragma unroll
        for (int t = 0; t < 2; ++t) {
            int idx = tid + t * 256;
            int kp = idx >> 5, d2 = idx & 31;
            unsigned int lo = (rv0[t] & 0xffffu) | (rv1[t] << 16);
            unsigned int hh = (rv0[t] >> 16) | (rv1[t] & 0xffff0000u);
            *(unsigned int*)&ldsVt[buf][2 * d2][2 * kp]     = lo;
            *(unsigned int*)&ldsVt[buf][2 * d2 + 1][2 * kp] = hh;
        }
    };

    stage_load(0);
    stage_store(0);
    __syncthreads();

    int cur = 0;
    for (int j = 0; j < NSEQ; j += 32) {
        bool more = (j + 32) < NSEQ;
        if (more) stage_load(j + 32);

        const unsigned short* Ks = &ldsK[cur][0][0];
        const unsigned short* Vs = &ldsVt[cur][0][0];

        v8f s0 = zero8(), s1 = zero8();
        v16bf kb0a = load_bfrag(Ks, 64, ln,       0, hi);
        v16bf kb0b = load_bfrag(Ks, 64, ln,      32, hi);
        v16bf kb1a = load_bfrag(Ks, 64, 16 + ln,  0, hi);
        v16bf kb1b = load_bfrag(Ks, 64, 16 + ln, 32, hi);
        s0 = wmma_bf16(qa0, kb0a, s0);
        s0 = wmma_bf16(qa1, kb0b, s0);
        s1 = wmma_bf16(qa0, kb1a, s1);
        s1 = wmma_bf16(qa1, kb1b, s1);

        // online softmax: row m = r + 8*hi lives in one 16-lane half
        float al[8];
#pragma unroll
        for (int r = 0; r < 8; ++r) {
            float mx = fmaxf(s0[r], s1[r]);
            mx = fmaxf(mx, __shfl_xor(mx, 1, 32));
            mx = fmaxf(mx, __shfl_xor(mx, 2, 32));
            mx = fmaxf(mx, __shfl_xor(mx, 4, 32));
            mx = fmaxf(mx, __shfl_xor(mx, 8, 32));
            float mn = fmaxf(mrow[r], mx);
            al[r] = __expf(mrow[r] - mn);
            float p0 = __expf(s0[r] - mn);
            float p1 = __expf(s1[r] - mn);
            s0[r] = p0; s1[r] = p1;
            float sum = p0 + p1;
            sum += __shfl_xor(sum, 1, 32);
            sum += __shfl_xor(sum, 2, 32);
            sum += __shfl_xor(sum, 4, 32);
            sum += __shfl_xor(sum, 8, 32);
            lrow[r] = lrow[r] * al[r] + sum;
            mrow[r] = mn;
        }

        // C-layout -> A-layout for P via wave-private LDS strip (in-order DS)
#pragma unroll
        for (int r = 0; r < 8; ++r) {
            int m = wave * 16 + r + hi * 8;
            ldsP[m][ln]      = f2bf(s0[r]);
            ldsP[m][16 + ln] = f2bf(s1[r]);
        }
        v16bf pa = load_afrag(&ldsP[0][0], 32, wave * 16 + ln, 0, hi);

#pragma unroll
        for (int dt = 0; dt < 4; ++dt) {
#pragma unroll
            for (int r = 0; r < 8; ++r) o[dt][r] *= al[r];
        }
#pragma unroll
        for (int dt = 0; dt < 4; ++dt) {
            v16bf vb = load_bfrag(Vs, 32, dt * 16 + ln, 0, hi);
            o[dt] = wmma_bf16(pa, vb, o[dt]);
        }

        if (more) stage_store(cur ^ 1);
        cur ^= 1;
        __syncthreads();
    }

    // epilogue: O /= l, merge heads -> Og[b*N + row, h*64 + d] bf16
#pragma unroll
    for (int dt = 0; dt < 4; ++dt) {
#pragma unroll
        for (int r = 0; r < 8; ++r) {
            int m = r + hi * 8;
            int row = q0 + wave * 16 + m;
            int col = h * HDIM + dt * 16 + ln;
            Og[((size_t)b * NSEQ + row) * DIMX + col] = f2bf(o[dt][r] / lrow[r]);
        }
    }
}

// ---------------------------------------------------------------------------
// Host-side launcher
// Inputs: x[4,2048,1024] f32, ln_gamma[1024], ln_beta[1024],
//   w_qkv[1024,3072] f32, w_out[1024,1024] f32.  Output: f32 [4,2048,1024].
// Workspace (bf16), ~92 MB total.
// ---------------------------------------------------------------------------
extern "C" void kernel_launch(void* const* d_in, const int* in_sizes, int n_in,
                              void* d_out, int out_size, void* d_ws, size_t ws_size,
                              hipStream_t stream) {
    const float* x     = (const float*)d_in[0];
    const float* gamma = (const float*)d_in[1];
    const float* beta  = (const float*)d_in[2];
    const float* wqkv  = (const float*)d_in[3];
    const float* wout  = (const float*)d_in[4];
    float* out = (float*)d_out;

    unsigned short* ws = (unsigned short*)d_ws;
    size_t off = 0;
    unsigned short* Xn = ws + off; off += (size_t)ROWS * DIMX;
    unsigned short* Wq = ws + off; off += (size_t)DIMX * NQKV;
    unsigned short* Wo = ws + off; off += (size_t)DIMX * DIMX;
    unsigned short* Qg = ws + off; off += (size_t)BATCH * HEADS * NSEQ * HDIM;
    unsigned short* Kg = ws + off; off += (size_t)BATCH * HEADS * NSEQ * HDIM;
    unsigned short* Vg = ws + off; off += (size_t)BATCH * HEADS * NSEQ * HDIM;
    unsigned short* AO = ws + off; off += (size_t)ROWS * DIMX;

    ln_bf16_kernel<<<ROWS, 256, 0, stream>>>(x, gamma, beta, Xn);

    {
        int n1 = DIMX * NQKV;
        cvt_bf16_kernel<<<(n1 + 255) / 256, 256, 0, stream>>>(wqkv, Wq, n1);
        int n2 = DIMX * DIMX;
        cvt_bf16_kernel<<<(n2 + 255) / 256, 256, 0, stream>>>(wout, Wo, n2);
    }

    {   // QKV GEMM: [8192,1024] x [1024,3072]
        dim3 grid(NQKV / 128, ROWS / 128);
        gemm_bf16_kernel<<<grid, 256, 0, stream>>>(Xn, Wq, ROWS, NQKV, DIMX,
                                                   0, nullptr, Qg, Kg, Vg);
    }

    {   // flash attention
        dim3 grid(NSEQ / 128, BATCH * HEADS);
        attn_kernel<<<grid, 256, 0, stream>>>(Qg, Kg, Vg, AO);
    }

    {   // out GEMM: [8192,1024] x [1024,1024] -> f32
        dim3 grid(DIMX / 128, ROWS / 128);
        gemm_bf16_kernel<<<grid, 256, 0, stream>>>(AO, Wo, ROWS, DIMX, DIMX,
                                                   1, out, nullptr, nullptr, nullptr);
    }
}
